// MultiModalFusionGAT_78666620994215
// MI455X (gfx1250) — matrix operand order repaired and checked by reading, not compiled, not run on hardware
//
#include <hip/hip_runtime.h>
#include <hip/hip_bf16.h>
#include <math.h>

// ---------------------------------------------------------------------------
// HGTConv forward for MI455X (gfx1250, wave32, WMMA).
// GEMMs: bf16 WMMA 16x16x32, f32 accumulate, 128x64 block tile,
//        double-buffered GLOBAL_LOAD_ASYNC_TO_LDS_B128 staging (ASYNCcnt).
// Edge softmax/aggregation: VALU + f32 atomics, b128 bf16 gathers.
// ---------------------------------------------------------------------------

typedef __bf16 bf16_t;
typedef bf16_t v16bf __attribute__((ext_vector_type(16)));
typedef float  v8f   __attribute__((ext_vector_type(8)));
typedef int    v4i   __attribute__((vector_size(16)));

#define NUM_USER 100000
#define NUM_ITEM 200000
#define NUM_TASTE 30000
#define NUM_INT 30000
#define NUM_IMG 30000
#define E_TI 200000
#define E_II 200000
#define E_GI 200000
#define E_UI 400000
#define E_IU 400000

#if defined(__has_builtin)
#if __has_builtin(__builtin_amdgcn_global_load_async_to_lds_b128) && \
    __has_builtin(__builtin_amdgcn_s_wait_asynccnt)
#define USE_ASYNC_LDS 1
#endif
#endif
#ifndef USE_ASYNC_LDS
#define USE_ASYNC_LDS 0
#endif

#if USE_ASYNC_LDS
__device__ __forceinline__ void async_b128(const void* g, void* l) {
  __builtin_amdgcn_global_load_async_to_lds_b128(
      (__attribute__((address_space(1))) v4i*)g,
      (__attribute__((address_space(3))) v4i*)l, 0, 0);
}
__device__ __forceinline__ void wait_async() {
  __builtin_amdgcn_s_wait_asynccnt(0);
}
#endif

// ordered-float encoding for atomicMax on f32 via uint
__device__ __forceinline__ unsigned fenc(float f) {
  unsigned u = __float_as_uint(f);
  return (int)u >= 0 ? (u | 0x80000000u) : ~u;
}
__device__ __forceinline__ float fdec(unsigned u) {
  return (u & 0x80000000u) ? __uint_as_float(u & 0x7FFFFFFFu)
                           : __uint_as_float(~u);
}

__device__ __forceinline__ float gelu_exact(float v) {
  return 0.5f * v * (1.0f + erff(v * 0.70710678118654752440f));
}

// unpack 8 packed bf16 (uint4) -> 8 f32 via bit shifts
__device__ __forceinline__ void unpack8(uint4 u, float* f) {
  f[0] = __uint_as_float(u.x << 16);
  f[1] = __uint_as_float(u.x & 0xFFFF0000u);
  f[2] = __uint_as_float(u.y << 16);
  f[3] = __uint_as_float(u.y & 0xFFFF0000u);
  f[4] = __uint_as_float(u.z << 16);
  f[5] = __uint_as_float(u.z & 0xFFFF0000u);
  f[6] = __uint_as_float(u.w << 16);
  f[7] = __uint_as_float(u.w & 0xFFFF0000u);
}

// ---------------------------------------------------------------------------
// KQV GEMM: out[M,384](bf16) = X[M,128](f32) @ W[128,384] + b
// Block 256 thr = 8 waves; block tile 128x64; wave: 16 rows x 4 col-subtiles.
// ---------------------------------------------------------------------------
__global__ __launch_bounds__(256)
void kqv_gemm(const float* __restrict__ X, const float* __restrict__ W,
              const float* __restrict__ bias, __hip_bfloat16* __restrict__ out,
              int M) {
  __shared__ alignas(16) float lds_a[2][128][36];
  __shared__ alignas(16) float lds_b[2][32][68];
  const int tid = threadIdx.x;
  const int lane = tid & 31, wave = tid >> 5;
  const int m16 = lane & 15, half = lane >> 4;
  const int blockRow = blockIdx.x * 128;
  const int col0 = blockIdx.y * 64;
  const bool full = (blockRow + 128) <= M;
  v8f acc[4];
#pragma unroll
  for (int c = 0; c < 4; ++c) acc[c] = {0.f, 0.f, 0.f, 0.f, 0.f, 0.f, 0.f, 0.f};

  auto compute = [&](int buf) {
    v16bf a;                               // A 16x32 bf16 lane layout (7.12.2)
    const int arow = wave * 16 + m16;
#pragma unroll
    for (int j = 0; j < 16; ++j) {
      int kA = (j < 8) ? (half * 8 + j) : (16 + half * 8 + (j - 8));
      a[j] = (bf16_t)lds_a[buf][arow][kA];
    }
#pragma unroll
    for (int c = 0; c < 4; ++c) {          // reuse A across 4 column subtiles
      v16bf b;
      const int bcol = c * 16 + m16;
#pragma unroll
      for (int j = 0; j < 16; ++j)
        b[j] = (bf16_t)lds_b[buf][(j < 8) ? (half * 8 + j) : (16 + half * 8 + (j - 8))][bcol];
      acc[c] = __builtin_amdgcn_wmma_f32_16x16x32_bf16(
          false, a, false, b, (short)0, acc[c], false, false);
    }
  };

  if (full) {
#if USE_ASYNC_LDS
    auto stage = [&](int kk0, int buf) {
      const float* Xb = X + (size_t)blockRow * 128 + kk0;
#pragma unroll
      for (int i = 0; i < 4; ++i) {        // A: 1024 b128 slots, 128 rows x 8
        int slot = tid + i * 256;
        int r = slot >> 3, c4 = slot & 7;
        async_b128(Xb + (size_t)r * 128 + c4 * 4, &lds_a[buf][r][c4 * 4]);
      }
#pragma unroll
      for (int i = 0; i < 2; ++i) {        // B 32x64: 512 b128 slots
        int slot = tid + i * 256;
        int kr = slot >> 4, c4 = slot & 15;
        async_b128(&W[(size_t)(kk0 + kr) * 384 + col0 + c4 * 4],
                   &lds_b[buf][kr][c4 * 4]);
      }
    };
    stage(0, 0);                           // prologue: chunk 0 -> buf 0
#pragma unroll
    for (int k = 0; k < 4; ++k) {
      wait_async();
      __syncthreads();                     // chunk k visible to all waves
      if (k < 3) stage((k + 1) * 32, (k + 1) & 1);  // prefetch behind WMMA
      compute(k & 1);
    }
#else
#pragma unroll
    for (int kk0 = 0; kk0 < 128; kk0 += 32) {
      const float4* Xv =
          reinterpret_cast<const float4*>(X + (size_t)blockRow * 128 + kk0);
#pragma unroll
      for (int i = 0; i < 4; ++i) {
        int slot = tid + i * 256;
        int r = slot >> 3, c4 = slot & 7;
        *reinterpret_cast<float4*>(&lds_a[0][r][c4 * 4]) =
            Xv[(size_t)r * 32 + c4];
      }
#pragma unroll
      for (int i = 0; i < 2; ++i) {
        int slot = tid + i * 256;
        int kr = slot >> 4, c4 = slot & 15;
        *reinterpret_cast<float4*>(&lds_b[0][kr][c4 * 4]) =
            *reinterpret_cast<const float4*>(
                &W[(size_t)(kk0 + kr) * 384 + col0 + c4 * 4]);
      }
      __syncthreads();
      compute(0);
      __syncthreads();
    }
#endif
  } else {  // guarded scalar path, last row-block only
    for (int kk0 = 0; kk0 < 128; kk0 += 32) {
#pragma unroll
      for (int i = 0; i < 16; ++i) {
        int idx = tid + i * 256;
        int r = idx >> 5, c = idx & 31;
        int row = blockRow + r;
        lds_a[0][r][c] = (row < M) ? X[(size_t)row * 128 + kk0 + c] : 0.f;
      }
#pragma unroll
      for (int i = 0; i < 2; ++i) {
        int slot = tid + i * 256;
        int kr = slot >> 4, c4 = slot & 15;
        *reinterpret_cast<float4*>(&lds_b[0][kr][c4 * 4]) =
            *reinterpret_cast<const float4*>(
                &W[(size_t)(kk0 + kr) * 384 + col0 + c4 * 4]);
      }
      __syncthreads();
      compute(0);
      __syncthreads();
    }
  }

#pragma unroll
  for (int c = 0; c < 4; ++c) {            // C layout: M = half*8 + r, N = m16
#pragma unroll
    for (int r = 0; r < 8; ++r) {
      int row = blockRow + wave * 16 + half * 8 + r;
      int col = col0 + c * 16 + m16;
      if (row < M)
        out[(size_t)row * 384 + col] = __float2bfloat16(acc[c][r] + bias[col]);
    }
  }
}

// ---------------------------------------------------------------------------
// HEAD GEMM: out = sig(skip)*(gelu(AGG) @ Wout + bout) + (1-sig)*x + x
// gelu applied at fragment-gather time so async staging stays a raw copy.
// ---------------------------------------------------------------------------
__global__ __launch_bounds__(256)
void head_gemm(const float* __restrict__ AGG, const float* __restrict__ W,
               const float* __restrict__ bias, const float* __restrict__ X,
               const float* __restrict__ skip, float* __restrict__ out, int M) {
  __shared__ alignas(16) float lds_a[2][128][36];
  __shared__ alignas(16) float lds_b[2][32][68];
  const int tid = threadIdx.x;
  const int lane = tid & 31, wave = tid >> 5;
  const int m16 = lane & 15, half = lane >> 4;
  const int blockRow = blockIdx.x * 128;
  const int col0 = blockIdx.y * 64;
  const bool full = (blockRow + 128) <= M;
  v8f acc[4];
#pragma unroll
  for (int c = 0; c < 4; ++c) acc[c] = {0.f, 0.f, 0.f, 0.f, 0.f, 0.f, 0.f, 0.f};

  auto compute = [&](int buf) {
    v16bf a;
    const int arow = wave * 16 + m16;
#pragma unroll
    for (int j = 0; j < 16; ++j) {
      int kA = (j < 8) ? (half * 8 + j) : (16 + half * 8 + (j - 8));
      a[j] = (bf16_t)gelu_exact(lds_a[buf][arow][kA]);
    }
#pragma unroll
    for (int c = 0; c < 4; ++c) {
      v16bf b;
      const int bcol = c * 16 + m16;
#pragma unroll
      for (int j = 0; j < 16; ++j)
        b[j] = (bf16_t)lds_b[buf][(j < 8) ? (half * 8 + j) : (16 + half * 8 + (j - 8))][bcol];
      acc[c] = __builtin_amdgcn_wmma_f32_16x16x32_bf16(
          false, a, false, b, (short)0, acc[c], false, false);
    }
  };

  if (full) {
#if USE_ASYNC_LDS
    auto stage = [&](int kk0, int buf) {
      const float* Ab = AGG + (size_t)blockRow * 128 + kk0;
#pragma unroll
      for (int i = 0; i < 4; ++i) {
        int slot = tid + i * 256;
        int r = slot >> 3, c4 = slot & 7;
        async_b128(Ab + (size_t)r * 128 + c4 * 4, &lds_a[buf][r][c4 * 4]);
      }
#pragma unroll
      for (int i = 0; i < 2; ++i) {
        int slot = tid + i * 256;
        int kr = slot >> 4, c4 = slot & 15;
        async_b128(&W[(size_t)(kk0 + kr) * 128 + col0 + c4 * 4],
                   &lds_b[buf][kr][c4 * 4]);
      }
    };
    stage(0, 0);
#pragma unroll
    for (int k = 0; k < 4; ++k) {
      wait_async();
      __syncthreads();
      if (k < 3) stage((k + 1) * 32, (k + 1) & 1);
      compute(k & 1);
    }
#else
#pragma unroll
    for (int kk0 = 0; kk0 < 128; kk0 += 32) {
      const float4* Av =
          reinterpret_cast<const float4*>(AGG + (size_t)blockRow * 128 + kk0);
#pragma unroll
      for (int i = 0; i < 4; ++i) {
        int slot = tid + i * 256;
        int r = slot >> 3, c4 = slot & 7;
        *reinterpret_cast<float4*>(&lds_a[0][r][c4 * 4]) =
            Av[(size_t)r * 32 + c4];
      }
#pragma unroll
      for (int i = 0; i < 2; ++i) {
        int slot = tid + i * 256;
        int kr = slot >> 4, c4 = slot & 15;
        *reinterpret_cast<float4*>(&lds_b[0][kr][c4 * 4]) =
            *reinterpret_cast<const float4*>(
                &W[(size_t)(kk0 + kr) * 128 + col0 + c4 * 4]);
      }
      __syncthreads();
      compute(0);
      __syncthreads();
    }
#endif
  } else {
    for (int kk0 = 0; kk0 < 128; kk0 += 32) {
#pragma unroll
      for (int i = 0; i < 16; ++i) {
        int idx = tid + i * 256;
        int r = idx >> 5, c = idx & 31;
        int row = blockRow + r;
        lds_a[0][r][c] = (row < M) ? AGG[(size_t)row * 128 + kk0 + c] : 0.f;
      }
#pragma unroll
      for (int i = 0; i < 2; ++i) {
        int slot = tid + i * 256;
        int kr = slot >> 4, c4 = slot & 15;
        *reinterpret_cast<float4*>(&lds_b[0][kr][c4 * 4]) =
            *reinterpret_cast<const float4*>(
                &W[(size_t)(kk0 + kr) * 128 + col0 + c4 * 4]);
      }
      __syncthreads();
      compute(0);
      __syncthreads();
    }
  }

  const float alpha = 1.0f / (1.0f + expf(-skip[0]));
#pragma unroll
  for (int c = 0; c < 4; ++c) {
#pragma unroll
    for (int r = 0; r < 8; ++r) {
      int row = blockRow + wave * 16 + half * 8 + r;
      int col = col0 + c * 16 + m16;
      if (row < M) {
        float og = acc[c][r] + bias[col];
        float xv = X[(size_t)row * 128 + col];
        out[(size_t)row * 128 + col] = alpha * og + (1.0f - alpha) * xv + xv;
      }
    }
  }
}

// ---------------------------------------------------------------------------
// Edge pass 1: score = (q_dst . (Wk[h] applied to k_src)) * p[h] / 4
// ---------------------------------------------------------------------------
__global__ __launch_bounds__(256)
void edge_score(const int* __restrict__ src, const int* __restrict__ dst, int E,
                const __hip_bfloat16* __restrict__ kqv_s,
                const __hip_bfloat16* __restrict__ kqv_d,
                const float* __restrict__ Wk, const float* __restrict__ p,
                float* __restrict__ score_out, unsigned* __restrict__ m_enc) {
  __shared__ float wk[2048];  // [8][16][16]
  for (int i = threadIdx.x; i < 2048; i += 256) wk[i] = Wk[i];
  __syncthreads();

  const int t = blockIdx.x * 256 + threadIdx.x;
  const int e = t >> 3, h = t & 7;
  if (e >= E) return;
  const int s = src[e], d = dst[e];
  const uint4* kp =
      reinterpret_cast<const uint4*>(kqv_s + (size_t)s * 384 + h * 16);
  const uint4* qp =
      reinterpret_cast<const uint4*>(kqv_d + (size_t)d * 384 + 128 + h * 16);
  float kv[16], qv[16];
  unpack8(kp[0], kv);
  unpack8(kp[1], kv + 8);
  unpack8(qp[0], qv);
  unpack8(qp[1], qv + 8);
  const float* wkh = wk + h * 256;
  float sc = 0.f;
#pragma unroll
  for (int o = 0; o < 16; ++o) {
    float kk = 0.f;
#pragma unroll
    for (int c = 0; c < 16; ++c) kk += kv[c] * wkh[c * 16 + o];
    sc += qv[o] * kk;
  }
  sc *= p[h] * 0.25f;  // 1/sqrt(D), D=16
  score_out[(size_t)e * 8 + h] = sc;
  atomicMax(m_enc + (size_t)d * 8 + h, fenc(sc));
}

// ---------------------------------------------------------------------------
// Edge pass 2: ex = exp(score - max); accumulate denominator.
// ---------------------------------------------------------------------------
__global__ __launch_bounds__(256)
void edge_exp(const int* __restrict__ dst, int E, float* __restrict__ score,
              const unsigned* __restrict__ m_enc, float* __restrict__ den) {
  const int t = blockIdx.x * 256 + threadIdx.x;
  if (t >= E * 8) return;
  const int e = t >> 3, h = t & 7;
  const int d = dst[e];
  const float m = fdec(m_enc[(size_t)d * 8 + h]);
  const float ex = expf(score[t] - m);
  score[t] = ex;
  atomicAdd(den + (size_t)d * 8 + h, ex);
}

// ---------------------------------------------------------------------------
// Edge pass 3: agg[d,h,:] += alpha * (Wv[h] applied to v_src)
// ---------------------------------------------------------------------------
__global__ __launch_bounds__(256)
void edge_scatter(const int* __restrict__ src, const int* __restrict__ dst,
                  int E, const __hip_bfloat16* __restrict__ kqv_s,
                  const float* __restrict__ Wv, const float* __restrict__ ex,
                  const float* __restrict__ den, float* __restrict__ agg) {
  __shared__ float wv[2048];
  for (int i = threadIdx.x; i < 2048; i += 256) wv[i] = Wv[i];
  __syncthreads();

  const int t = blockIdx.x * 256 + threadIdx.x;
  const int e = t >> 3, h = t & 7;
  if (e >= E) return;
  const int s = src[e], d = dst[e];
  const float alpha = ex[(size_t)e * 8 + h] / den[(size_t)d * 8 + h];
  const uint4* vp =
      reinterpret_cast<const uint4*>(kqv_s + (size_t)s * 384 + 256 + h * 16);
  float vv[16];
  unpack8(vp[0], vv);
  unpack8(vp[1], vv + 8);
  const float* wvh = wv + h * 256;
  float* ag = agg + (size_t)d * 128 + h * 16;
#pragma unroll
  for (int o = 0; o < 16; ++o) {
    float m = 0.f;
#pragma unroll
    for (int c = 0; c < 16; ++c) m += vv[c] * wvh[c * 16 + o];
    atomicAdd(ag + o, m * alpha);
  }
}

// ---------------------------------------------------------------------------
extern "C" void kernel_launch(void* const* d_in, const int* in_sizes, int n_in,
                              void* d_out, int out_size, void* d_ws,
                              size_t ws_size, hipStream_t stream) {
  // ---- inputs, setup_inputs() dict order ----
  const float* x_user = (const float*)d_in[0];
  const float* x_item = (const float*)d_in[1];
  const float* x_taste = (const float*)d_in[2];
  const float* x_int = (const float*)d_in[3];
  const float* x_img = (const float*)d_in[4];
  const int* src_ti = (const int*)d_in[5];
  const int* dst_ti = (const int*)d_in[6];
  const int* src_ii = (const int*)d_in[7];
  const int* dst_ii = (const int*)d_in[8];
  const int* src_gi = (const int*)d_in[9];
  const int* dst_gi = (const int*)d_in[10];
  const int* src_ui = (const int*)d_in[11];
  const int* dst_ui = (const int*)d_in[12];
  const int* src_iu = (const int*)d_in[13];
  const int* dst_iu = (const int*)d_in[14];
  const float* Wkqv_user = (const float*)d_in[15];
  const float* bkqv_user = (const float*)d_in[16];
  const float* Wkqv_item = (const float*)d_in[17];
  const float* bkqv_item = (const float*)d_in[18];
  const float* Wkqv_taste = (const float*)d_in[19];
  const float* bkqv_taste = (const float*)d_in[20];
  const float* Wkqv_int = (const float*)d_in[21];
  const float* bkqv_int = (const float*)d_in[22];
  const float* Wkqv_img = (const float*)d_in[23];
  const float* bkqv_img = (const float*)d_in[24];
  const float* Wout_user = (const float*)d_in[25];
  const float* bout_user = (const float*)d_in[26];
  const float* skip_user = (const float*)d_in[27];
  const float* Wout_item = (const float*)d_in[28];
  const float* bout_item = (const float*)d_in[29];
  const float* skip_item = (const float*)d_in[30];
  const float* Wk_ti = (const float*)d_in[31];
  const float* Wv_ti = (const float*)d_in[32];
  const float* p_ti = (const float*)d_in[33];
  const float* Wk_ii = (const float*)d_in[34];
  const float* Wv_ii = (const float*)d_in[35];
  const float* p_ii = (const float*)d_in[36];
  const float* Wk_gi = (const float*)d_in[37];
  const float* Wv_gi = (const float*)d_in[38];
  const float* p_gi = (const float*)d_in[39];
  const float* Wk_ui = (const float*)d_in[40];
  const float* Wv_ui = (const float*)d_in[41];
  const float* p_ui = (const float*)d_in[42];
  const float* Wk_iu = (const float*)d_in[43];
  const float* Wv_iu = (const float*)d_in[44];
  const float* p_iu = (const float*)d_in[45];

  // ---- workspace carve (~520 MB) ----
  char* wp = (char*)d_ws;
  auto carve = [&](size_t bytes) -> void* {
    void* r = (void*)wp;
    wp += (bytes + 255) & ~(size_t)255;
    return r;
  };
  __hip_bfloat16* kqvU = (__hip_bfloat16*)carve((size_t)NUM_USER * 384 * 2);
  __hip_bfloat16* kqvI = (__hip_bfloat16*)carve((size_t)NUM_ITEM * 384 * 2);
  __hip_bfloat16* kqvT = (__hip_bfloat16*)carve((size_t)NUM_TASTE * 384 * 2);
  __hip_bfloat16* kqvN = (__hip_bfloat16*)carve((size_t)NUM_INT * 384 * 2);
  __hip_bfloat16* kqvG = (__hip_bfloat16*)carve((size_t)NUM_IMG * 384 * 2);
  float* exI = (float*)carve((size_t)(E_TI + E_II + E_GI + E_UI) * 8 * 4);
  float* exU = (float*)carve((size_t)E_IU * 8 * 4);
  char* zero_begin = wp;
  unsigned* mI = (unsigned*)carve((size_t)NUM_ITEM * 8 * 4);
  float* denI = (float*)carve((size_t)NUM_ITEM * 8 * 4);
  unsigned* mU = (unsigned*)carve((size_t)NUM_USER * 8 * 4);
  float* denU = (float*)carve((size_t)NUM_USER * 8 * 4);
  float* aggI = (float*)carve((size_t)NUM_ITEM * 128 * 4);
  float* aggU = (float*)carve((size_t)NUM_USER * 128 * 4);
  (void)hipMemsetAsync(zero_begin, 0, (size_t)(wp - zero_begin), stream);

  const dim3 blk(256);

  // ---- fused KQV projections (bf16 WMMA GEMMs) ----
  kqv_gemm<<<dim3((NUM_USER + 127) / 128, 6), blk, 0, stream>>>(
      x_user, Wkqv_user, bkqv_user, kqvU, NUM_USER);
  kqv_gemm<<<dim3((NUM_ITEM + 127) / 128, 6), blk, 0, stream>>>(
      x_item, Wkqv_item, bkqv_item, kqvI, NUM_ITEM);
  kqv_gemm<<<dim3((NUM_TASTE + 127) / 128, 6), blk, 0, stream>>>(
      x_taste, Wkqv_taste, bkqv_taste, kqvT, NUM_TASTE);
  kqv_gemm<<<dim3((NUM_INT + 127) / 128, 6), blk, 0, stream>>>(
      x_int, Wkqv_int, bkqv_int, kqvN, NUM_INT);
  kqv_gemm<<<dim3((NUM_IMG + 127) / 128, 6), blk, 0, stream>>>(
      x_img, Wkqv_img, bkqv_img, kqvG, NUM_IMG);

  auto eb = [](int E) { return (E * 8 + 255) / 256; };
  const size_t oTI = 0, oII = (size_t)E_TI * 8, oGI = (size_t)(E_TI + E_II) * 8,
               oUI = (size_t)(E_TI + E_II + E_GI) * 8;

  // ---- pass 1: scores + segment max (joint softmax across item relations) --
  edge_score<<<eb(E_TI), blk, 0, stream>>>(src_ti, dst_ti, E_TI, kqvT, kqvI,
                                           Wk_ti, p_ti, exI + oTI, mI);
  edge_score<<<eb(E_II), blk, 0, stream>>>(src_ii, dst_ii, E_II, kqvN, kqvI,
                                           Wk_ii, p_ii, exI + oII, mI);
  edge_score<<<eb(E_GI), blk, 0, stream>>>(src_gi, dst_gi, E_GI, kqvG, kqvI,
                                           Wk_gi, p_gi, exI + oGI, mI);
  edge_score<<<eb(E_UI), blk, 0, stream>>>(src_ui, dst_ui, E_UI, kqvU, kqvI,
                                           Wk_ui, p_ui, exI + oUI, mI);
  edge_score<<<eb(E_IU), blk, 0, stream>>>(src_iu, dst_iu, E_IU, kqvI, kqvU,
                                           Wk_iu, p_iu, exU, mU);

  // ---- pass 2: exp + denominator ----
  edge_exp<<<eb(E_TI), blk, 0, stream>>>(dst_ti, E_TI, exI + oTI, mI, denI);
  edge_exp<<<eb(E_II), blk, 0, stream>>>(dst_ii, E_II, exI + oII, mI, denI);
  edge_exp<<<eb(E_GI), blk, 0, stream>>>(dst_gi, E_GI, exI + oGI, mI, denI);
  edge_exp<<<eb(E_UI), blk, 0, stream>>>(dst_ui, E_UI, exI + oUI, mI, denI);
  edge_exp<<<eb(E_IU), blk, 0, stream>>>(dst_iu, E_IU, exU, mU, denU);

  // ---- pass 3: weighted value scatter ----
  edge_scatter<<<eb(E_TI), blk, 0, stream>>>(src_ti, dst_ti, E_TI, kqvT, Wv_ti,
                                             exI + oTI, denI, aggI);
  edge_scatter<<<eb(E_II), blk, 0, stream>>>(src_ii, dst_ii, E_II, kqvN, Wv_ii,
                                             exI + oII, denI, aggI);
  edge_scatter<<<eb(E_GI), blk, 0, stream>>>(src_gi, dst_gi, E_GI, kqvG, Wv_gi,
                                             exI + oGI, denI, aggI);
  edge_scatter<<<eb(E_UI), blk, 0, stream>>>(src_ui, dst_ui, E_UI, kqvU, Wv_ui,
                                             exI + oUI, denI, aggI);
  edge_scatter<<<eb(E_IU), blk, 0, stream>>>(src_iu, dst_iu, E_IU, kqvI, Wv_iu,
                                             exU, denU, aggU);

  // ---- output heads (WMMA GEMM + gelu + skip mix + residual) ----
  float* out_user = (float*)d_out;
  float* out_item = out_user + (size_t)NUM_USER * 128;
  float* out_taste = out_item + (size_t)NUM_ITEM * 128;
  float* out_int = out_taste + (size_t)NUM_TASTE * 128;
  float* out_img = out_int + (size_t)NUM_INT * 128;

  head_gemm<<<dim3((NUM_USER + 127) / 128, 2), blk, 0, stream>>>(
      aggU, Wout_user, bout_user, x_user, skip_user, out_user, NUM_USER);
  head_gemm<<<dim3((NUM_ITEM + 127) / 128, 2), blk, 0, stream>>>(
      aggI, Wout_item, bout_item, x_item, skip_item, out_item, NUM_ITEM);

  // ---- passthrough node types ----
  (void)hipMemcpyAsync(out_taste, x_taste, (size_t)NUM_TASTE * 128 * 4,
                       hipMemcpyDeviceToDevice, stream);
  (void)hipMemcpyAsync(out_int, x_int, (size_t)NUM_INT * 128 * 4,
                       hipMemcpyDeviceToDevice, stream);
  (void)hipMemcpyAsync(out_img, x_img, (size_t)NUM_IMG * 128 * 4,
                       hipMemcpyDeviceToDevice, stream);
}
